// Net_60808146976930
// MI455X (gfx1250) — matrix operand order repaired and checked by reading, not compiled, not run on hardware
//
#include <hip/hip_runtime.h>

// ---------------------------------------------------------------------------
// GNN MetaLayer x2 + edge head for MI455X (gfx1250), bf16 WMMA (16x16x32).
//
//   - weights prepacked (fp32 -> bf16) into exact B-fragment lane order
//   - gathers (x[src], x[dst]) folded into the A-fragment row address
//   - concat(...)@W decomposed into sums of partial GEMMs (no concat buffers)
//   - segment_sum fused into the GEMM epilogue via global f32 atomics
//   - layer-1 node model skipped (its output is unused by the head)
//   - 2x4 register tiling per wave: 8 independent accumulators ->
//     8-deep independent WMMA stream (hides WMMA hazard slots) and
//     24 B of loads per WMMA (vs 64 B untiled / 32 B at 2x2)
// ---------------------------------------------------------------------------

typedef __attribute__((ext_vector_type(16))) __bf16 v16bf;
typedef __attribute__((ext_vector_type(8)))  __bf16 v8bf;
typedef __attribute__((ext_vector_type(8)))  float  v8f;

static constexpr int NN     = 10000;   // nodes
static constexpr int NN_PAD = 10016;   // padded to a multiple of 32 for GEMM M
static constexpr int NE     = 160000;  // edges (multiple of 32)

// ---------------------------- small utility kernels ------------------------

__global__ void f32_to_bf16_k(const float* __restrict__ in, __bf16* __restrict__ out, int n) {
  int i = blockIdx.x * 256 + threadIdx.x;
  if (i < n) out[i] = (__bf16)in[i];
}

// edge_attr [E,6] fp32 -> bf16 padded to [E,32] (zeros beyond col 6)
__global__ void pad_ea_k(const float* __restrict__ ea, __bf16* __restrict__ out, int ecnt) {
  int t = blockIdx.x * 256 + threadIdx.x;
  if (t >= ecnt * 32) return;
  int e = t >> 5, j = t & 31;
  out[t] = (__bf16)((j < 6) ? ea[e * 6 + j] : 0.0f);
}

__global__ void zero_f32_k(float* __restrict__ p, int n) {
  int i = blockIdx.x * 256 + threadIdx.x;
  if (i < n) p[i] = 0.0f;
}

__global__ void degree_k(const int* __restrict__ dst, float* __restrict__ cnt, int ecnt) {
  int e = blockIdx.x * 256 + threadIdx.x;
  if (e < ecnt) atomicAdd(&cnt[dst[e]], 1.0f);
}

// s[N,256] f32 -> bf16, divided by max(cnt,1)  (mean aggregation)
__global__ void mean_bf16_k(const float* __restrict__ s, const float* __restrict__ cnt,
                            __bf16* __restrict__ sb, int n) {
  int t = blockIdx.x * 256 + threadIdx.x;
  if (t >= n) return;
  float c = fmaxf(cnt[t >> 8], 1.0f);
  sb[t] = (__bf16)(s[t] / c);
}

// Prepack fp32 weight rows [rowOffset .. rowOffset+Kvalid) x 256 into bf16
// WMMA B-fragment order: out[((ct*kchunks + kc)*32 + lane)*16 + i]
//   element (i): K = kc*32 + (lane>>4)*16 + i, N = ct*16 + (lane&15)
// Zero-pads K beyond Kvalid (handles F_EDGE=6 and 1030-row matrices).
__global__ void prepack_k(const float* __restrict__ W, int rowOffset, int Kvalid,
                          int kchunks, __bf16* __restrict__ out) {
  int t = blockIdx.x * 256 + threadIdx.x;
  int total = 16 * kchunks * 32;
  if (t >= total) return;
  int lane = t & 31, rem = t >> 5;
  int kc = rem % kchunks, ct = rem / kchunks;
  int n = ct * 16 + (lane & 15);
  int kbase = kc * 32 + (lane >> 4) * 16;
  __bf16* d = out + (size_t)t * 16;
#pragma unroll
  for (int i = 0; i < 16; ++i) {
    int k = kbase + i;
    d[i] = (__bf16)((k < Kvalid) ? W[(size_t)(rowOffset + k) * 256 + n] : 0.0f);
  }
}

// ------------------------------ WMMA GEMM core -----------------------------

__device__ __forceinline__ v16bf make_a(v8bf lo, v8bf hi) {
  v16bf a;
#pragma unroll
  for (int i = 0; i < 8; ++i) { a[i] = lo[i]; a[8 + i] = hi[i]; }
  return a;
}

#define WMMA_BF16(A, B, C) \
  __builtin_amdgcn_wmma_f32_16x16x32_bf16(false, (A), false, (B), (short)0, (C), false, false)

// Accumulate one source into the wave's 2x4 tile block:
//   acc[rt][ctl] += gather(A, idx)[rows rt] @ Wpacked[:, col tile ct0+ctl]
// A-fragment in contiguous k order (same as the prepacked B):
//   lane m = lane&15 holds row M=m; kh = lane>>4 selects K halves:
//   a[0..15] = A[row][k0 + kh*16 + i]       (one 32B coalesced load)
// B-fragment: 16 contiguous halves per lane from the prepacked buffer (32B).
// Per k-chunk: 192 B of loads feed 8 independent WMMAs (24 B/WMMA).
__device__ __forceinline__ void acc_src(v8f (&acc)[2][4],
                                        const __bf16* A, const int* idx, int lda,
                                        const __bf16* W, int kchunks,
                                        int blockRow, int lane, int ct0) {
  if (A == nullptr) return;
  int m  = lane & 15;
  int kh = lane >> 4;
  int row0 = blockRow * 32 + m;
  int row1 = row0 + 16;
  int r0 = idx ? idx[row0] : row0;
  int r1 = idx ? idx[row1] : row1;
  const __bf16* Ar0 = A + (size_t)r0 * lda;
  const __bf16* Ar1 = A + (size_t)r1 * lda;
  const __bf16* Wt0 = W + ((size_t)ct0 * kchunks * 512 + (size_t)lane * 16);
  const size_t wstride = (size_t)kchunks * 512;  // halves per column tile
  for (int kc = 0; kc < kchunks; ++kc) {
    int ko = kc * 32 + kh * 16;
    v16bf a0 = *(const v16bf*)(Ar0 + ko);
    v16bf a1 = *(const v16bf*)(Ar1 + ko);
    const __bf16* wp = Wt0 + (size_t)kc * 512;
    v16bf b0 = *(const v16bf*)(wp);
    v16bf b1 = *(const v16bf*)(wp + wstride);
    v16bf b2 = *(const v16bf*)(wp + 2 * wstride);
    v16bf b3 = *(const v16bf*)(wp + 3 * wstride);
    acc[0][0] = WMMA_BF16(a0, b0, acc[0][0]);
    acc[0][1] = WMMA_BF16(a0, b1, acc[0][1]);
    acc[0][2] = WMMA_BF16(a0, b2, acc[0][2]);
    acc[0][3] = WMMA_BF16(a0, b3, acc[0][3]);
    acc[1][0] = WMMA_BF16(a1, b0, acc[1][0]);
    acc[1][1] = WMMA_BF16(a1, b1, acc[1][1]);
    acc[1][2] = WMMA_BF16(a1, b2, acc[1][2]);
    acc[1][3] = WMMA_BF16(a1, b3, acc[1][3]);
  }
}

// Generic fused MLP-layer GEMM:
//   Y[M x 256] = act( sum_s gather(A_s, idx_s) @ W_s + bias ) (+ residual)
// Output either stored as bf16, or scatter-added (f32 atomics) at scatterIdx.
// Block = 128 threads (4 waves); wave w owns a 32x64 output patch:
// rows = blockIdx.x*32 .. +31, col tiles {4w .. 4w+3}. Grid = (M/32). M%32==0.
__global__ __launch_bounds__(128)
void gemm_wmma_k(const __bf16* A0, const int* idx0, int lda0, const __bf16* W0, int kc0,
                 const __bf16* A1, const int* idx1, int lda1, const __bf16* W1, int kc1,
                 const __bf16* A2, const int* idx2, int lda2, const __bf16* W2, int kc2,
                 const float* __restrict__ bias,
                 const __bf16* __restrict__ residual,
                 __bf16* __restrict__ outB,
                 float* __restrict__ outAtomic, const int* __restrict__ scatterIdx,
                 int doRelu) {
  int lane = threadIdx.x & 31;
  int wave = threadIdx.x >> 5;
  int blockRow = blockIdx.x;
  int ct0 = wave * 4;

  v8f acc[2][4] = {};
  acc_src(acc, A0, idx0, lda0, W0, kc0, blockRow, lane, ct0);
  acc_src(acc, A1, idx1, lda1, W1, kc1, blockRow, lane, ct0);
  acc_src(acc, A2, idx2, lda2, W2, kc2, blockRow, lane, ct0);

  // C/D layout: VGPR r -> M = r + (lane>=16 ? 8 : 0), N = lane&15
  int nl = lane & 15;
  int mb = (lane >> 4) * 8;
#pragma unroll
  for (int ctl = 0; ctl < 4; ++ctl) {
    int n = (ct0 + ctl) * 16 + nl;
    float bv = bias ? bias[n] : 0.0f;
#pragma unroll
    for (int rt = 0; rt < 2; ++rt) {
#pragma unroll
      for (int r = 0; r < 8; ++r) {
        int gm = blockRow * 32 + rt * 16 + mb + r;
        float v = acc[rt][ctl][r] + bv;
        if (doRelu) v = fmaxf(v, 0.0f);
        if (residual) v += (float)residual[(size_t)gm * 256 + n];
        if (outAtomic) {  // fused segment_sum: scatter-add at destination node
          atomicAdd(&outAtomic[(size_t)scatterIdx[gm] * 256 + n], v);
        } else {
          outB[(size_t)gm * 256 + n] = (__bf16)v;
        }
      }
    }
  }
}

// Final head: out[e] = dot(h[e,:256], p_w2) + p_b2
__global__ void head_k(const __bf16* __restrict__ h, const float* __restrict__ w2,
                       const float* __restrict__ b2, float* __restrict__ out, int ecnt) {
  int e = blockIdx.x * 256 + threadIdx.x;
  if (e >= ecnt) return;
  const v8bf* hv = (const v8bf*)(h + (size_t)e * 256);
  float acc = 0.0f;
  for (int j = 0; j < 32; ++j) {
    v8bf hb = hv[j];
#pragma unroll
    for (int i = 0; i < 8; ++i) acc += (float)hb[i] * w2[j * 8 + i];
  }
  out[e] = acc + b2[0];
}

// --------------------------------- launch ----------------------------------

extern "C" void kernel_launch(void* const* d_in, const int* in_sizes, int n_in,
                              void* d_out, int out_size, void* d_ws, size_t ws_size,
                              hipStream_t stream) {
  (void)in_sizes; (void)n_in; (void)out_size; (void)ws_size;

  const float* x    = (const float*)d_in[0];
  const int*   eidx = (const int*)d_in[1];
  const float* ea   = (const float*)d_in[2];
  const int* src = eidx;        // edge_index[0]
  const int* dst = eidx + NE;   // edge_index[1]
  float* out = (float*)d_out;

  // ---- workspace carve-out (256B aligned) ----
  char* base = (char*)d_ws;
  size_t off = 0;
  auto alloc = [&](size_t bytes) -> void* {
    void* r = base + off;
    off = (off + bytes + 255) & ~(size_t)255;
    return r;
  };
  // Node-side buffers padded to NN_PAD rows so node GEMMs can run M=NN_PAD
  // (padded rows hold garbage; src/dst indices are < NN so it is never used).
  __bf16* xb   = (__bf16*)alloc((size_t)NN_PAD * 512 * 2);  // x in bf16
  __bf16* eap  = (__bf16*)alloc((size_t)NE * 32 * 2);       // edge_attr padded bf16
  __bf16* bufE = (__bf16*)alloc((size_t)NE * 256 * 2);      // eh / eh1 / h (reused)
  __bf16* ea1  = (__bf16*)alloc((size_t)NE * 256 * 2);      // layer0 edge output
  __bf16* ea2  = (__bf16*)alloc((size_t)NE * 256 * 2);      // layer1 edge output
  float*  sAgg = (float*) alloc((size_t)NN * 256 * 4);      // segment sum (f32)
  float*  cnt  = (float*) alloc((size_t)NN * 4);            // in-degree
  __bf16* sb   = (__bf16*)alloc((size_t)NN_PAD * 256 * 2);  // mean-agg bf16
  __bf16* nh   = (__bf16*)alloc((size_t)NN_PAD * 256 * 2);
  __bf16* x1   = (__bf16*)alloc((size_t)NN_PAD * 256 * 2);  // layer0 node output
  // prepacked bf16 weights (fragment order); bytes = kchunks * 16384
  auto wbuf = [&](int kchunks) { return (__bf16*)alloc((size_t)kchunks * 16384); };
  __bf16* pw_e1a = wbuf(16);  // e_w1_0 rows   0:512  (x_i)
  __bf16* pw_e1b = wbuf(16);  // e_w1_0 rows 512:1024 (x_j)
  __bf16* pw_e1c = wbuf(1);   // e_w1_0 rows 1024:1030 pad->32 (ea)
  __bf16* pw_e20 = wbuf(8);   // e_w2_0
  __bf16* pw_m0a = wbuf(16);  // n_wm_0 rows 0:512  (x_i)
  __bf16* pw_m0b = wbuf(8);   // n_wm_0 rows 512:768 (ea_new)
  __bf16* pw_n1a = wbuf(16);  // n_w1_0 rows 0:512  (x)
  __bf16* pw_n1b = wbuf(8);   // n_w1_0 rows 512:768 (s)
  __bf16* pw_n20 = wbuf(8);   // n_w2_0
  __bf16* pw_f1a = wbuf(8);   // e_w1_1 rows 0:256   (x1_i)
  __bf16* pw_f1b = wbuf(8);   // e_w1_1 rows 256:512 (x1_j)
  __bf16* pw_f1c = wbuf(8);   // e_w1_1 rows 512:768 (ea1)
  __bf16* pw_e21 = wbuf(8);   // e_w2_1
  __bf16* pw_p1  = wbuf(8);   // p_w1

  dim3 blk256(256), blk128(128);
  auto pp = [&](const void* W, int rowOff, int Kvalid, int kchunks, __bf16* o) {
    int total = 16 * kchunks * 32;
    prepack_k<<<(total + 255) / 256, blk256, 0, stream>>>((const float*)W, rowOff, Kvalid, kchunks, o);
  };
  auto gemm = [&](int M,  // must be a multiple of 32
                  const __bf16* A0, const int* i0, int l0, const __bf16* W0, int k0,
                  const __bf16* A1, const int* i1, int l1, const __bf16* W1, int k1,
                  const __bf16* A2, const int* i2, int l2, const __bf16* W2, int k2,
                  const float* bias, const __bf16* res, __bf16* oB,
                  float* oAt, const int* scat, int relu) {
    dim3 grid(M / 32);
    gemm_wmma_k<<<grid, blk128, 0, stream>>>(A0, i0, l0, W0, k0, A1, i1, l1, W1, k1,
                                             A2, i2, l2, W2, k2, bias, res, oB, oAt, scat, relu);
  };

  // ---- stage 0: conversions / prepack / zero-init ----
  f32_to_bf16_k<<<((size_t)NN * 512 + 255) / 256, blk256, 0, stream>>>(x, xb, NN * 512);
  pad_ea_k<<<((size_t)NE * 32 + 255) / 256, blk256, 0, stream>>>(ea, eap, NE);
  zero_f32_k<<<((size_t)NN * 256 + 255) / 256, blk256, 0, stream>>>(sAgg, NN * 256);
  zero_f32_k<<<(NN + 255) / 256, blk256, 0, stream>>>(cnt, NN);
  pp(d_in[3],    0,  512, 16, pw_e1a);
  pp(d_in[3],  512,  512, 16, pw_e1b);
  pp(d_in[3], 1024,    6,  1, pw_e1c);
  pp(d_in[5],    0,  256,  8, pw_e20);
  pp(d_in[7],    0,  512, 16, pw_m0a);
  pp(d_in[7],  512,  256,  8, pw_m0b);
  pp(d_in[9],    0,  512, 16, pw_n1a);
  pp(d_in[9],  512,  256,  8, pw_n1b);
  pp(d_in[11],   0,  256,  8, pw_n20);
  pp(d_in[13],   0,  256,  8, pw_f1a);
  pp(d_in[13], 256,  256,  8, pw_f1b);
  pp(d_in[13], 512,  256,  8, pw_f1c);
  pp(d_in[15],   0,  256,  8, pw_e21);
  pp(d_in[23],   0,  256,  8, pw_p1);

  // ---- layer 0 edge model: eh = relu([x_i,x_j,ea]@e_w1_0 + b); ea1 = eh@e_w2_0 + b
  gemm(NE, xb, src, 512, pw_e1a, 16,
           xb, dst, 512, pw_e1b, 16,
           eap, nullptr, 32, pw_e1c, 1,
       (const float*)d_in[4], nullptr, bufE, nullptr, nullptr, 1);
  gemm(NE, bufE, nullptr, 256, pw_e20, 8,
           nullptr, nullptr, 0, nullptr, 0, nullptr, nullptr, 0, nullptr, 0,
       (const float*)d_in[6], nullptr, ea1, nullptr, nullptr, 0);

  // ---- layer 0 node model: msg = relu([x_i,ea1]@n_wm_0 + b); scatter-add at dst
  gemm(NE, xb, src, 512, pw_m0a, 16,
           ea1, nullptr, 256, pw_m0b, 8,
           nullptr, nullptr, 0, nullptr, 0,
       (const float*)d_in[8], nullptr, nullptr, sAgg, dst, 1);
  degree_k<<<(NE + 255) / 256, blk256, 0, stream>>>(dst, cnt, NE);
  mean_bf16_k<<<((size_t)NN * 256 + 255) / 256, blk256, 0, stream>>>(sAgg, cnt, sb, NN * 256);
  // nh = relu([x,s]@n_w1_0 + b); x1 = nh@n_w2_0 + b   (M padded to 10016)
  gemm(NN_PAD, xb, nullptr, 512, pw_n1a, 16,
               sb, nullptr, 256, pw_n1b, 8,
               nullptr, nullptr, 0, nullptr, 0,
       (const float*)d_in[10], nullptr, nh, nullptr, nullptr, 1);
  gemm(NN_PAD, nh, nullptr, 256, pw_n20, 8,
               nullptr, nullptr, 0, nullptr, 0, nullptr, nullptr, 0, nullptr, 0,
       (const float*)d_in[12], nullptr, x1, nullptr, nullptr, 0);

  // ---- layer 1 edge model (node model output is unused downstream -> skipped)
  gemm(NE, x1, src, 256, pw_f1a, 8,
           x1, dst, 256, pw_f1b, 8,
           ea1, nullptr, 256, pw_f1c, 8,
       (const float*)d_in[14], nullptr, bufE, nullptr, nullptr, 1);
  gemm(NE, bufE, nullptr, 256, pw_e21, 8,
           nullptr, nullptr, 0, nullptr, 0, nullptr, nullptr, 0, nullptr, 0,
       (const float*)d_in[16], ea1 /*residual*/, ea2, nullptr, nullptr, 0);

  // ---- head: h = relu(ea2@p_w1 + b); out = h@p_w2 + b
  gemm(NE, ea2, nullptr, 256, pw_p1, 8,
           nullptr, nullptr, 0, nullptr, 0, nullptr, nullptr, 0, nullptr, 0,
       (const float*)d_in[24], nullptr, bufE, nullptr, nullptr, 1);
  head_k<<<(NE + 255) / 256, blk256, 0, stream>>>(bufE, (const float*)d_in[25],
                                                  (const float*)d_in[26], out, NE);
}
